// HGT_1013612282097
// MI455X (gfx1250) — matrix operand order repaired and checked by reading, not compiled, not run on hardware
//
#include <hip/hip_runtime.h>
#include <math.h>

// ---------------------------------------------------------------------------
// Problem constants (from reference)
// ---------------------------------------------------------------------------
#define NS0 80000
#define NS1 60000
#define NS2 30000
#define NTOT 170000
#define OFF1 80000
#define OFF2 140000
#define NLAYER 4
#define NB 64

typedef __attribute__((ext_vector_type(16))) __bf16 v16bf;
typedef __attribute__((ext_vector_type(8)))  float  v8f;

#if __has_builtin(__builtin_amdgcn_global_load_async_to_lds_b128)
#define HAVE_ASYNC_LDS 1
typedef int v4i __attribute__((vector_size(16)));
typedef __attribute__((address_space(1))) v4i* gv4p;   // global (AS1) v4i*
typedef __attribute__((address_space(3))) v4i* sv4p;   // LDS (AS3) v4i*
#endif

// ---------------------------------------------------------------------------
// Helpers
// ---------------------------------------------------------------------------
__device__ __forceinline__ unsigned short f2bf(float f) {
  unsigned u = __float_as_uint(f);
  unsigned r = u + 0x7FFFu + ((u >> 16) & 1u);   // round-to-nearest-even
  return (unsigned short)(r >> 16);
}
__device__ __forceinline__ unsigned pack2bf(float a, float b) {
  return (unsigned)f2bf(a) | ((unsigned)f2bf(b) << 16);
}
__device__ __forceinline__ float geluf(float x) {
  float x3 = x * x * x;
  return 0.5f * x * (1.0f + tanhf(0.7978845608028654f * (x + 0.044715f * x3)));
}
// Order-preserving float<->uint encoding for atomic max
__device__ __forceinline__ unsigned encF(float f) {
  unsigned u = __float_as_uint(f);
  return (u & 0x80000000u) ? ~u : (u | 0x80000000u);
}
__device__ __forceinline__ float decF(unsigned k) {
  return (k & 0x80000000u) ? __uint_as_float(k ^ 0x80000000u) : __uint_as_float(~k);
}
#define ENC_NEG_INF 0x007FFFFFu   // encF(-inf)

// ---------------------------------------------------------------------------
// One-time weight transpose + f32->bf16: Wt[N][K] = bf16(W[K][N])
// K, N are multiples of 32 for every weight matrix in this model.
// ---------------------------------------------------------------------------
__global__ __launch_bounds__(256)
void wtrans(const float* __restrict__ W, unsigned short* __restrict__ Wt, int K, int N)
{
  __shared__ unsigned short tile[32][33];
  int bk = blockIdx.x * 32, bn = blockIdx.y * 32;
  int tx = threadIdx.x & 31, ty = threadIdx.x >> 5;   // 32 x 8
  #pragma unroll
  for (int r = 0; r < 32; r += 8)
    tile[ty + r][tx] = f2bf(W[(size_t)(bk + ty + r) * N + bn + tx]);  // coalesced in n
  __syncthreads();
  #pragma unroll
  for (int r = 0; r < 32; r += 8)
    Wt[(size_t)(bn + ty + r) * K + bk + tx] = tile[tx][ty + r];       // coalesced in k
}

// ---------------------------------------------------------------------------
// Tiled WMMA GEMM:  C[M,N] = epilogue( act(A[M,K]) @ W[K,N] + bias )
//  - A: f32 [M,K] (lda), converted to bf16 while staging to LDS
//  - Wt: bf16, pre-transposed [N][K]
//  - Block tile 128x128, 256 threads = 8 waves (4 M x 2 N), wave tile 32x64
//  - preactGelu: gelu(A) while staging;  skipPtr: C = s*acc + (1-s)*Hm
// ---------------------------------------------------------------------------
#define TM 128
#define TN 128
#define TK 32
#define AST 40   // LDS stride in halves, A row-major [m][k]
#define BST 40   // LDS stride in halves, B transposed [n][k]

__global__ __launch_bounds__(256)
void gemm_wmma(const float* __restrict__ A, int lda,
               const unsigned short* __restrict__ Wt,
               const float* __restrict__ bias,
               float* __restrict__ C, int ldc,
               int M, int K, int preactGelu,
               const float* __restrict__ skipPtr,
               const float* __restrict__ Hm, int ldh)
{
  __shared__ __align__(16) unsigned short As[TM * AST];
  __shared__ __align__(16) unsigned short Bs[TN * BST];

  const int t    = threadIdx.x;
  const int rm0  = blockIdx.x * TM;
  const int cn0  = blockIdx.y * TN;
  const int wave = t >> 5, lane = t & 31;
  const int wm = wave >> 1, wn = wave & 1;       // 4 x 2 wave grid
  const int mlane = lane & 15, hi = lane >> 4;

  v8f zero = {0.f,0.f,0.f,0.f,0.f,0.f,0.f,0.f};
  v8f acc[2][4];
  #pragma unroll
  for (int i = 0; i < 2; ++i)
    #pragma unroll
    for (int j = 0; j < 4; ++j) acc[i][j] = zero;

  for (int kt = 0; kt < K; kt += TK) {
    // ---- stage A tile 128x32 (f32 -> bf16), row-major [m][k]
    #pragma unroll
    for (int i = 0; i < 4; ++i) {
      int idx = t + i * 256;
      int row = idx >> 3;
      int kq  = (idx & 7) << 2;
      int gr  = rm0 + row;
      float4 v = make_float4(0.f, 0.f, 0.f, 0.f);
      if (gr < M) v = *(const float4*)(A + (size_t)gr * lda + kt + kq);
      if (preactGelu) { v.x = geluf(v.x); v.y = geluf(v.y); v.z = geluf(v.z); v.w = geluf(v.w); }
      unsigned* d32 = (unsigned*)&As[row * AST + kq];
      d32[0] = pack2bf(v.x, v.y);
      d32[1] = pack2bf(v.z, v.w);
    }
    // ---- stage B tile 128(n) x 32(k) bf16, already [n][k] in memory
    #pragma unroll
    for (int i = 0; i < 2; ++i) {
      int idx = t + i * 256;
      int n  = idx >> 2;
      int kq = (idx & 3) << 3;
      const unsigned short* gp = Wt + (size_t)(cn0 + n) * K + kt + kq;
      unsigned short* lp = &Bs[n * BST + kq];
#ifdef HAVE_ASYNC_LDS
      __builtin_amdgcn_global_load_async_to_lds_b128((gv4p)(void*)gp, (sv4p)(void*)lp, 0, 0);
#else
      *(uint4*)lp = *(const uint4*)gp;
#endif
    }
#ifdef HAVE_ASYNC_LDS
#if __has_builtin(__builtin_amdgcn_s_wait_asynccnt)
    __builtin_amdgcn_s_wait_asynccnt(0);
#else
    asm volatile("s_wait_asynccnt 0" ::: "memory");
#endif
#endif
    __syncthreads();

    // ---- WMMA fragments (layouts per CDNA5 ISA 7.12.2)
    v16bf af[2], bf[4];
    #pragma unroll
    for (int sm = 0; sm < 2; ++sm) {
      int row = wm * 32 + sm * 16 + mlane;
      int kb  = hi * 8;                    // A: lane group hi holds k kb..kb+7, kb+16..kb+23
      union { uint4 q[2]; v16bf v; } u;
      u.q[0] = *(const uint4*)&As[row * AST + kb];
      u.q[1] = *(const uint4*)&As[row * AST + kb + 16];
      af[sm] = u.v;
    }
    #pragma unroll
    for (int sn = 0; sn < 4; ++sn) {
      int n   = wn * 64 + sn * 16 + mlane;
      int kb2 = hi * 8;                    // B: lane group hi holds k kb2..kb2+7, kb2+16..kb2+23
      union { uint4 q[2]; v16bf v; } u;
      u.q[0] = *(const uint4*)&Bs[n * BST + kb2];
      u.q[1] = *(const uint4*)&Bs[n * BST + kb2 + 16];
      bf[sn] = u.v;
    }
    #pragma unroll
    for (int sm = 0; sm < 2; ++sm)
      #pragma unroll
      for (int sn = 0; sn < 4; ++sn)
        acc[sm][sn] = __builtin_amdgcn_wmma_f32_16x16x32_bf16(
            false, af[sm], false, bf[sn], (short)0, acc[sm][sn], false, false);
    __syncthreads();
  }

  // ---- epilogue
  float alv = 1.f, omalv = 0.f;
  if (skipPtr) { float s = *skipPtr; alv = 1.f / (1.f + expf(-s)); omalv = 1.f - alv; }
  #pragma unroll
  for (int sm = 0; sm < 2; ++sm) {
    #pragma unroll
    for (int sn = 0; sn < 4; ++sn) {
      int col = cn0 + wn * 64 + sn * 16 + mlane;
      float bv = bias ? bias[col] : 0.f;
      #pragma unroll
      for (int r = 0; r < 8; ++r) {
        int row = rm0 + wm * 32 + sm * 16 + hi * 8 + r;   // D: lane group hi -> M = hi*8 + r
        if (row < M) {
          float cv = acc[sm][sn][r] + bv;
          if (skipPtr) cv = alv * cv + omalv * Hm[(size_t)row * ldh + col];
          C[(size_t)row * ldc + col] = cv;
        }
      }
    }
  }
}

// ---------------------------------------------------------------------------
// LayerNorm over 256 features; one wave per node; params selected by node type
// ---------------------------------------------------------------------------
__global__ __launch_bounds__(256)
void ln_kernel(const float* __restrict__ in, int inStride,
               const float* __restrict__ gArr, const float* __restrict__ bArr,
               float* __restrict__ out, int n1, int n2, int ntot)
{
  int wave = threadIdx.x >> 5, lane = threadIdx.x & 31;
  int n = blockIdx.x * 8 + wave;
  if (n >= ntot) return;
  int type = (n >= n2) ? 2 : ((n >= n1) ? 1 : 0);
  const float* x = in + (size_t)n * inStride;
  float v[8];
  float4 a = *(const float4*)(x + lane * 8);
  float4 b = *(const float4*)(x + lane * 8 + 4);
  v[0]=a.x; v[1]=a.y; v[2]=a.z; v[3]=a.w; v[4]=b.x; v[5]=b.y; v[6]=b.z; v[7]=b.w;
  float s = 0.f;
  #pragma unroll
  for (int j = 0; j < 8; ++j) s += v[j];
  for (int o = 16; o; o >>= 1) s += __shfl_xor(s, o, 32);
  float m = s * (1.0f / 256.0f);
  float q = 0.f;
  #pragma unroll
  for (int j = 0; j < 8; ++j) { float d = v[j] - m; q += d * d; }
  for (int o = 16; o; o >>= 1) q += __shfl_xor(q, o, 32);
  float rstd = rsqrtf(q * (1.0f / 256.0f) + 1e-5f);
  const float* g  = gArr + type * 256;
  const float* bb = bArr + type * 256;
  float* o = out + (size_t)n * 256;
  #pragma unroll
  for (int j = 0; j < 8; ++j) {
    int f = lane * 8 + j;
    o[f] = (v[j] - m) * rstd * g[f] + bb[f];
  }
}

// ---------------------------------------------------------------------------
// Attention scratch init
// ---------------------------------------------------------------------------
__global__ void init_attn(unsigned* smax, float* ssum, float* aggr, int ntot)
{
  int i = blockIdx.x * blockDim.x + threadIdx.x;
  if (i < ntot * 4) { smax[i] = ENC_NEG_INF; ssum[i] = 0.f; }
  if (i < ntot * 256) aggr[i] = 0.f;
}

// ---------------------------------------------------------------------------
// Edge pass A: per-edge attention logits + atomic segment-max (per dst, head)
// ---------------------------------------------------------------------------
__global__ void edge_attn_a(const int* __restrict__ ei, int ne,
                            const float* __restrict__ kqv, int dstOff,
                            const float* __restrict__ kr,
                            const float* __restrict__ prel,
                            float* __restrict__ ea, unsigned* __restrict__ smax)
{
  int e = blockIdx.x * blockDim.x + threadIdx.x;
  if (e >= ne) return;
  int src = ei[e], dst = ei[ne + e];
  int gd = dstOff + dst;
  const float* q = kqv + (size_t)gd * 768 + 256;
  const float* k = kr + (size_t)src * 256;
  #pragma unroll
  for (int h = 0; h < 4; ++h) {
    float dot = 0.f;
    for (int d = 0; d < 64; d += 4) {
      float4 qa = *(const float4*)(q + h * 64 + d);
      float4 ka = *(const float4*)(k + h * 64 + d);
      dot += qa.x * ka.x + qa.y * ka.y + qa.z * ka.z + qa.w * ka.w;
    }
    float a = dot * prel[h] * 0.125f;   // / sqrt(64)
    ea[(size_t)e * 4 + h] = a;
    atomicMax(&smax[(size_t)gd * 4 + h], encF(a));
  }
}

// ---------------------------------------------------------------------------
// Edge pass B: exp weights, segment-sum of weights, numerator scatter-add
// ---------------------------------------------------------------------------
__global__ void edge_aggr_b(const int* __restrict__ ei, int ne,
                            const float* __restrict__ vr, int dstOff,
                            const float* __restrict__ ea,
                            const unsigned* __restrict__ smax,
                            float* __restrict__ ssum, float* __restrict__ aggr)
{
  int e = blockIdx.x * blockDim.x + threadIdx.x;
  if (e >= ne) return;
  int src = ei[e], dst = ei[ne + e];
  int gd = dstOff + dst;
  float w[4];
  #pragma unroll
  for (int h = 0; h < 4; ++h) {
    w[h] = expf(ea[(size_t)e * 4 + h] - decF(smax[(size_t)gd * 4 + h]));
    atomicAdd(&ssum[(size_t)gd * 4 + h], w[h]);
  }
  const float* v = vr + (size_t)src * 256;
  float* dstp = aggr + (size_t)gd * 256;
  for (int f = 0; f < 256; f += 4) {
    float4 vv = *(const float4*)(v + f);
    float ww = w[f >> 6];
    atomicAdd(&dstp[f + 0], vv.x * ww);
    atomicAdd(&dstp[f + 1], vv.y * ww);
    atomicAdd(&dstp[f + 2], vv.z * ww);
    atomicAdd(&dstp[f + 3], vv.w * ww);
  }
}

// aggr /= (z + 1e-16)  per (node, head)
__global__ void normalize_aggr(float* __restrict__ aggr,
                               const float* __restrict__ ssum, int ntot)
{
  int i = blockIdx.x * blockDim.x + threadIdx.x;
  if (i >= ntot * 256) return;
  int n = i >> 8, f = i & 255;
  aggr[i] /= (ssum[(size_t)n * 4 + (f >> 6)] + 1e-16f);
}

// ---------------------------------------------------------------------------
// Gated pooling
// ---------------------------------------------------------------------------
__global__ void init_gate(unsigned* gmax, float* gsum, float* emb)
{
  int i = blockIdx.x * blockDim.x + threadIdx.x;
  if (i < 192) { gmax[i] = ENC_NEG_INF; gsum[i] = 0.f; }
  if (i < NB * 784) { if ((i % 784) < 768) emb[i] = 0.f; }
}

__global__ void gate_score(const float* __restrict__ X,
                           const float* __restrict__ wg, const float* __restrict__ bg,
                           const int* __restrict__ b0, const int* __restrict__ b1,
                           const int* __restrict__ b2,
                           float* __restrict__ gs, unsigned* __restrict__ gmax)
{
  int n = blockIdx.x * blockDim.x + threadIdx.x;
  if (n >= NTOT) return;
  int type, g;
  if (n < OFF1)      { type = 0; g = b0[n]; }
  else if (n < OFF2) { type = 1; g = b1[n - OFF1]; }
  else               { type = 2; g = b2[n - OFF2]; }
  const float* x = X + (size_t)n * 256;
  const float* w = wg + type * 256;
  float s = bg[type];
  for (int f = 0; f < 256; f += 4) {
    float4 xv = *(const float4*)(x + f);
    float4 wv = *(const float4*)(w + f);
    s += xv.x * wv.x + xv.y * wv.y + xv.z * wv.z + xv.w * wv.w;
  }
  gs[n] = s;
  atomicMax(&gmax[type * 64 + g], encF(s));
}

__global__ void gate_pool(const float* __restrict__ X,
                          const int* __restrict__ b0, const int* __restrict__ b1,
                          const int* __restrict__ b2,
                          const float* __restrict__ gs, const unsigned* __restrict__ gmax,
                          float* __restrict__ gsum, float* __restrict__ emb)
{
  int n = blockIdx.x * blockDim.x + threadIdx.x;
  if (n >= NTOT) return;
  int type, g;
  if (n < OFF1)      { type = 0; g = b0[n]; }
  else if (n < OFF2) { type = 1; g = b1[n - OFF1]; }
  else               { type = 2; g = b2[n - OFF2]; }
  float w = expf(gs[n] - decF(gmax[type * 64 + g]));
  atomicAdd(&gsum[type * 64 + g], w);
  const float* x = X + (size_t)n * 256;
  float* d = emb + (size_t)g * 784 + type * 256;
  for (int f = 0; f < 256; ++f) atomicAdd(&d[f], w * x[f]);
}

__global__ void gate_norm(float* __restrict__ emb, const float* __restrict__ gsum)
{
  int i = blockIdx.x * blockDim.x + threadIdx.x;
  if (i >= NB * 768) return;
  int g = i / 768, c = i % 768;
  int type = c >> 8;
  emb[(size_t)g * 784 + c] /= (gsum[type * 64 + g] + 1e-16f);
}

// y_base MLP -> emb cols 768..783
__global__ void hy_kernel(const float* __restrict__ y,
                          const float* __restrict__ w1, const float* __restrict__ c1,
                          const float* __restrict__ w2, const float* __restrict__ c2,
                          float* __restrict__ emb)
{
  int g = threadIdx.x;
  if (g >= NB) return;
  float yv = y[g];
  float h1[16];
  #pragma unroll
  for (int j = 0; j < 16; ++j) {
    float t = yv * w1[j] + c1[j];
    h1[j] = t > 0.f ? t : 0.2f * t;
  }
  #pragma unroll
  for (int j = 0; j < 16; ++j) {
    float t = c2[j];
    #pragma unroll
    for (int k = 0; k < 16; ++k) t += h1[k] * w2[k * 16 + j];
    emb[(size_t)g * 784 + 768 + j] = t;
  }
}

// dense -> batchnorm (over 64 rows) -> gelu ; one block per output column
__global__ __launch_bounds__(64)
void mlp_bn_gelu(const float* __restrict__ in, int K,
                 const float* __restrict__ wg, const float* __restrict__ bg,
                 const float* __restrict__ gamma, const float* __restrict__ beta,
                 float* __restrict__ out, int N)
{
  int j = blockIdx.x, r = threadIdx.x;
  float acc = bg[j];
  for (int k = 0; k < K; ++k) acc += in[(size_t)r * K + k] * wg[(size_t)k * N + j];
  __shared__ float red[64];
  red[r] = acc;
  __syncthreads();
  for (int o = 32; o; o >>= 1) { if (r < o) red[r] += red[r + o]; __syncthreads(); }
  float m = red[0] * (1.0f / 64.0f);
  __syncthreads();
  red[r] = (acc - m) * (acc - m);
  __syncthreads();
  for (int o = 32; o; o >>= 1) { if (r < o) red[r] += red[r + o]; __syncthreads(); }
  float var = red[0] * (1.0f / 64.0f);
  float xn = (acc - m) * rsqrtf(var + 1e-5f) * gamma[j] + beta[j];
  out[(size_t)r * N + j] = geluf(xn);
}

__global__ void final_out(const float* __restrict__ o2, const float* __restrict__ w3,
                          const float* __restrict__ b3, float* __restrict__ out)
{
  int g = threadIdx.x;
  if (g >= NB) return;
  float a = b3[0];
  for (int k = 0; k < 128; ++k) a += o2[(size_t)g * 128 + k] * w3[k];
  out[g] = a;
}

// ---------------------------------------------------------------------------
// Host orchestration
// ---------------------------------------------------------------------------
static inline int cdiv(int a, int b) { return (a + b - 1) / b; }

static void gemm(hipStream_t s, const float* A, int lda,
                 const unsigned short* Wt, int N, int K,
                 const float* bias, float* C, int ldc, int M,
                 int preact, const float* skipP, const float* Hm, int ldh)
{
  dim3 grid(cdiv(M, TM), N / TN);
  gemm_wmma<<<grid, 256, 0, s>>>(A, lda, Wt, bias, C, ldc, M, K, preact, skipP, Hm, ldh);
}

extern "C" void kernel_launch(void* const* d_in, const int* in_sizes, int n_in,
                              void* d_out, int out_size, void* d_ws, size_t ws_size,
                              hipStream_t stream)
{
  (void)in_sizes; (void)n_in; (void)out_size; (void)ws_size;
  const float* x[3]    = {(const float*)d_in[0], (const float*)d_in[1], (const float*)d_in[2]};
  const float* y_base  = (const float*)d_in[3];
  const float* W_in    = (const float*)d_in[4];
  const float* b_in    = (const float*)d_in[5];
  const float* ln_g    = (const float*)d_in[6];
  const float* ln_b    = (const float*)d_in[7];
  const float* W_kqv   = (const float*)d_in[8];
  const float* b_kqv   = (const float*)d_in[9];
  const float* W_krel  = (const float*)d_in[10];
  const float* W_vrel  = (const float*)d_in[11];
  const float* p_rel   = (const float*)d_in[12];
  const float* W_out   = (const float*)d_in[13];
  const float* b_out   = (const float*)d_in[14];
  const float* skip    = (const float*)d_in[15];
  const float* W_jk    = (const float*)d_in[16];
  const float* b_jk    = (const float*)d_in[17];
  const float* W_gate  = (const float*)d_in[18];
  const float* b_gate  = (const float*)d_in[19];
  const float* W_y1    = (const float*)d_in[20];
  const float* b_y1    = (const float*)d_in[21];
  const float* W_y2    = (const float*)d_in[22];
  const float* b_y2    = (const float*)d_in[23];
  const float* Wg1     = (const float*)d_in[24];
  const float* bg1     = (const float*)d_in[25];
  const float* g1      = (const float*)d_in[26];
  const float* beta1   = (const float*)d_in[27];
  const float* Wg2     = (const float*)d_in[28];
  const float* bg2     = (const float*)d_in[29];
  const float* g2      = (const float*)d_in[30];
  const float* beta2   = (const float*)d_in[31];
  const float* Wg3     = (const float*)d_in[32];
  const float* bg3     = (const float*)d_in[33];
  const int*   ei[4]   = {(const int*)d_in[34], (const int*)d_in[35],
                          (const int*)d_in[36], (const int*)d_in[37]};
  const int*   batch[3]= {(const int*)d_in[38], (const int*)d_in[39], (const int*)d_in[40]};

  const int NS[3]   = {NS0, NS1, NS2};
  const int off[3]  = {0, OFF1, OFF2};
  const int NE[4]   = {320000, 320000, 160000, 160000};
  const int etSrc[4]= {0, 1, 0, 2};
  const int etDst[4]= {1, 0, 2, 0};
  const int vrOff[4]= {0, 80000, 140000, 220000};   // row offsets in VR buffer
  const int eaOff[4]= {0, 320000, 640000, 800000};  // edge offsets in EA buffer

  // ---- workspace layout (floats); ~2.1 GB + 12.3 MB bf16 weights
  float* ws = (float*)d_ws;
  size_t p = 0;
  float*    X0   = ws + p; p += (size_t)NTOT * 256;   // per-layer input / JK-proj out
  float*    JK   = ws + p; p += (size_t)NTOT * 1024;  // per-layer outputs, concat layout
  float*    HB   = ws + p; p += (size_t)NTOT * 256;   // LN output
  float*    KQV  = ws + p; p += (size_t)NTOT * 768;
  float*    KR   = ws + p; p += (size_t)80000 * 256;  // reused per edge type
  float*    VR   = ws + p; p += (size_t)250000 * 256; // per-edge-type segments
  float*    AGG  = ws + p; p += (size_t)NTOT * 256;
  float*    EA   = ws + p; p += (size_t)960000 * 4;
  unsigned* SMAX = (unsigned*)(ws + p); p += (size_t)NTOT * 4;
  float*    SSUM = ws + p; p += (size_t)NTOT * 4;
  float*    GS   = ws + p; p += (size_t)NTOT;
  unsigned* GMAX = (unsigned*)(ws + p); p += 192;
  float*    GSUM = ws + p; p += 192;
  float*    EMB  = ws + p; p += (size_t)NB * 784;
  float*    O1   = ws + p; p += (size_t)NB * 256;
  float*    O2   = ws + p; p += (size_t)NB * 128;
  // bf16 pre-transposed weights [N][K]
  unsigned short* WT     = (unsigned short*)(ws + p);
  unsigned short* WTin   = WT;                        // 3 x [256][128]
  unsigned short* WTkqv  = WTin   + (size_t)3  * 32768;   // 12 x [768][256]
  unsigned short* WTkrel = WTkqv  + (size_t)12 * 196608;  // 16 x [256][256]
  unsigned short* WTvrel = WTkrel + (size_t)16 * 65536;   // 16 x [256][256]
  unsigned short* WTout  = WTvrel + (size_t)16 * 65536;   // 12 x [256][256]
  unsigned short* WTjk   = WTout  + (size_t)12 * 65536;   // 3 x [256][1024]

  // ---- one-time weight transpose/convert (graph-capture safe, deterministic)
  for (int i = 0; i < 3; ++i)
    wtrans<<<dim3(4, 8),  256, 0, stream>>>(W_in   + (size_t)i * 32768,  WTin   + (size_t)i * 32768,  128, 256);
  for (int i = 0; i < 12; ++i)
    wtrans<<<dim3(8, 24), 256, 0, stream>>>(W_kqv  + (size_t)i * 196608, WTkqv  + (size_t)i * 196608, 256, 768);
  for (int i = 0; i < 16; ++i)
    wtrans<<<dim3(8, 8),  256, 0, stream>>>(W_krel + (size_t)i * 65536,  WTkrel + (size_t)i * 65536,  256, 256);
  for (int i = 0; i < 16; ++i)
    wtrans<<<dim3(8, 8),  256, 0, stream>>>(W_vrel + (size_t)i * 65536,  WTvrel + (size_t)i * 65536,  256, 256);
  for (int i = 0; i < 12; ++i)
    wtrans<<<dim3(8, 8),  256, 0, stream>>>(W_out  + (size_t)i * 65536,  WTout  + (size_t)i * 65536,  256, 256);
  for (int i = 0; i < 3; ++i)
    wtrans<<<dim3(32, 8), 256, 0, stream>>>(W_jk   + (size_t)i * 262144, WTjk   + (size_t)i * 262144, 1024, 256);

  // ---- input projection: X0 = x_i @ W_in[i] + b_in[i]
  for (int i = 0; i < 3; ++i)
    gemm(stream, x[i], 128, WTin + (size_t)i * 32768, 256, 128,
         b_in + i * 256, X0 + (size_t)off[i] * 256, 256, NS[i],
         0, nullptr, nullptr, 0);

  // ---- transformer layers
  for (int l = 0; l < NLAYER; ++l) {
    const float* inPtr = (l == 0) ? X0 : (JK + (size_t)(l - 1) * 256);
    int          inStr = (l == 0) ? 256 : 1024;

    ln_kernel<<<cdiv(NTOT, 8), 256, 0, stream>>>(
        inPtr, inStr, ln_g + (size_t)l * 3 * 256, ln_b + (size_t)l * 3 * 256,
        HB, OFF1, OFF2, NTOT);

    for (int i = 0; i < 3; ++i)
      gemm(stream, HB + (size_t)off[i] * 256, 256,
           WTkqv + (size_t)(l * 3 + i) * 196608, 768, 256,
           b_kqv + (size_t)(l * 3 + i) * 768,
           KQV + (size_t)off[i] * 768, 768, NS[i], 0, nullptr, nullptr, 0);

    init_attn<<<cdiv(NTOT * 256, 256), 256, 0, stream>>>(SMAX, SSUM, AGG, NTOT);

    for (int e = 0; e < 4; ++e) {
      int s = etSrc[e];
      gemm(stream, KQV + (size_t)off[s] * 768, 768,
           WTkrel + (size_t)(l * 4 + e) * 65536, 256, 256, nullptr,
           KR, 256, NS[s], 0, nullptr, nullptr, 0);
      gemm(stream, KQV + (size_t)off[s] * 768 + 512, 768,
           WTvrel + (size_t)(l * 4 + e) * 65536, 256, 256, nullptr,
           VR + (size_t)vrOff[e] * 256, 256, NS[s], 0, nullptr, nullptr, 0);
      edge_attn_a<<<cdiv(NE[e], 256), 256, 0, stream>>>(
          ei[e], NE[e], KQV, off[etDst[e]], KR,
          p_rel + (size_t)(l * 4 + e) * 4, EA + (size_t)eaOff[e] * 4, SMAX);
    }
    for (int e = 0; e < 4; ++e)
      edge_aggr_b<<<cdiv(NE[e], 256), 256, 0, stream>>>(
          ei[e], NE[e], VR + (size_t)vrOff[e] * 256, off[etDst[e]],
          EA + (size_t)eaOff[e] * 4, SMAX, SSUM, AGG);

    normalize_aggr<<<cdiv(NTOT * 256, 256), 256, 0, stream>>>(AGG, SSUM, NTOT);

    // out proj with fused gelu pre-activation + sigmoid-skip mix, into JK col block l
    for (int i = 0; i < 3; ++i)
      gemm(stream, AGG + (size_t)off[i] * 256, 256,
           WTout + (size_t)(l * 3 + i) * 65536, 256, 256,
           b_out + (size_t)(l * 3 + i) * 256,
           JK + (size_t)off[i] * 1024 + (size_t)l * 256, 1024, NS[i],
           1, skip + (l * 3 + i), HB + (size_t)off[i] * 256, 256);
  }

  // ---- jumping-knowledge projection: X0 = JK @ W_jk[i] + b_jk[i]
  for (int i = 0; i < 3; ++i)
    gemm(stream, JK + (size_t)off[i] * 1024, 1024,
         WTjk + (size_t)i * 262144, 256, 1024, b_jk + i * 256,
         X0 + (size_t)off[i] * 256, 256, NS[i], 0, nullptr, nullptr, 0);

  // ---- gated pooling into EMB[64, 784]
  init_gate<<<cdiv(NB * 784, 256), 256, 0, stream>>>(GMAX, GSUM, EMB);
  gate_score<<<cdiv(NTOT, 256), 256, 0, stream>>>(X0, W_gate, b_gate,
      batch[0], batch[1], batch[2], GS, GMAX);
  gate_pool<<<cdiv(NTOT, 256), 256, 0, stream>>>(X0, batch[0], batch[1], batch[2],
      GS, GMAX, GSUM, EMB);
  gate_norm<<<cdiv(NB * 768, 256), 256, 0, stream>>>(EMB, GSUM);
  hy_kernel<<<1, 64, 0, stream>>>(y_base, W_y1, b_y1, W_y2, b_y2, EMB);

  // ---- head: dense + BN + gelu (x2), final dense
  mlp_bn_gelu<<<256, 64, 0, stream>>>(EMB, 784, Wg1, bg1, g1, beta1, O1, 256);
  mlp_bn_gelu<<<128, 64, 0, stream>>>(O1, 256, Wg2, bg2, g2, beta2, O2, 128);
  final_out<<<1, 64, 0, stream>>>(O2, Wg3, bg3, (float*)d_out);
}
